// MLA_KVSplice_41815801594363
// MI455X (gfx1250) — matrix operand order repaired and checked by reading, not compiled, not run on hardware
//
#include <hip/hip_runtime.h>
#include <math.h>
#include <stdint.h>

// ---------------------------------------------------------------- problem dims
#define T_TOK   2048
#define D_MODEL 2048
#define N_HEADS 16
#define HEAD_DIM 128
#define D_LAT   512
#define D_CMP   256
#define N_KV    4096          // 2*H*HD
#define SM_SCALE 0.08838834764831845f   // 1/sqrt(128)

// ---------------------------------------------------------------- WMMA types
typedef __bf16 bf16;
typedef __attribute__((ext_vector_type(16))) __bf16 bf16x16;
typedef __attribute__((ext_vector_type(8)))  __bf16 bf16x8;
typedef __attribute__((ext_vector_type(8)))  float  f32x8;
typedef uint32_t u32x4 __attribute__((ext_vector_type(4)));
typedef uint32_t u32x8 __attribute__((ext_vector_type(8)));

union V16 { bf16x16 v; bf16x8 h[2]; };

__device__ __forceinline__ f32x8 wmma_bf16(bf16x16 a, bf16x16 b, f32x8 c) {
  // (neg_a, A, neg_b, B, c_mod, C, reuse_a, reuse_b)
  return __builtin_amdgcn_wmma_f32_16x16x32_bf16(false, a, false, b, (short)0, c,
                                                 false, false);
}

// ---------------------------------------------------------------- TDM helper
// Issue a 2D tile load (bf16 elements) via the Tensor Data Mover.
// tile_w elements per row, tile_h rows, source row pitch = pitch_elems.
// LDS destination gets 4 DWORDs of padding after every 16 DWORDs (i.e. rows of
// 32 bf16 are stored with stride 40 bf16 = LDT).
__device__ __forceinline__ void tdm_load_tile_bf16(
    uint32_t lds_addr, const bf16* gptr,
    uint32_t rem_elems0, uint32_t rem_rows1, uint32_t pitch_elems,
    uint32_t tile_w, uint32_t tile_h)
{
  const uint64_t ga = (uint64_t)(uintptr_t)gptr;
  u32x4 g0;
  g0[0] = 1u;                                                 // count=1, user D#
  g0[1] = __builtin_amdgcn_readfirstlane(lds_addr);           // lds_addr
  g0[2] = __builtin_amdgcn_readfirstlane((uint32_t)ga);       // global_addr lo
  g0[3] = __builtin_amdgcn_readfirstlane(
              ((uint32_t)(ga >> 32) & 0x01FFFFFFu) | (2u << 30)); // hi | type=2
  u32x8 g1;
  // data_size=1 (2B), pad_enable=1, pad_interval=3 (16 DW), pad_amount=3 (4 DW)
  g1[0] = (1u << 16) | (1u << 20) | (3u << 22) | (3u << 25);
  g1[1] = (rem_elems0 & 0xFFFFu) << 16;                       // tensor_dim0 lo
  g1[2] = (rem_elems0 >> 16) | ((rem_rows1 & 0xFFFFu) << 16); // dim0 hi|dim1 lo
  g1[3] = (rem_rows1 >> 16) | (tile_w << 16);                 // dim1 hi|tile_dim0
  g1[4] = tile_h;                                             // tile_dim1
  g1[5] = pitch_elems;                                        // dim0_stride lo
  g1[6] = 0u;                                                 // stride hi / dim1_stride lo
  g1[7] = 0u;
  asm volatile("tensor_load_to_lds %0, %1" :: "s"(g0), "s"(g1) : "memory");
}

// ---------------------------------------------------------------- bf16 GEMM
// C[M,N] (f32) = A[M,K] (bf16, row major) @ B[N,K]^T (bf16, row major) + bias
// M % 128 == 0, N % 128 == 0, K % 32 == 0.
#define BM 128
#define BN 128
#define BK 32
#define LDT (BK + 8)

__global__ __launch_bounds__(256) void gemm_bf16_kernel(
    const bf16* __restrict__ A, const bf16* __restrict__ B,
    const float* __restrict__ bias, float* __restrict__ C,
    int M, int N, int K)
{
  __shared__ bf16 As[BM][LDT];
  __shared__ bf16 Bs[BN][LDT];

  const int tid  = threadIdx.x;
  const int lane = tid & 31, wid = tid >> 5;
  const int lh   = lane >> 4, lm = lane & 15;       // half-wave, lane-in-half
  const int wm   = (wid & 3) * 32;                  // wave row offset in tile
  const int wn   = (wid >> 2) * 64;                 // wave col offset in tile
  const int bm   = blockIdx.y * BM, bn = blockIdx.x * BN;

  const uint32_t ldsA = (uint32_t)(uintptr_t)&As[0][0];
  const uint32_t ldsB = (uint32_t)(uintptr_t)&Bs[0][0];

  f32x8 acc[2][4] = {};
  const int ktn = K / BK;

  for (int kt = 0; kt < ktn; ++kt) {
    const int k0 = kt * BK;
    // ---- TDM-staged tiles: wave 0 issues two tensor loads, everyone waits.
    if (wid == 0) {
      tdm_load_tile_bf16(ldsA, A + (size_t)bm * K + k0,
                         (uint32_t)(K - k0), (uint32_t)(M - bm),
                         (uint32_t)K, BK, BM);
      tdm_load_tile_bf16(ldsB, B + (size_t)bn * K + k0,
                         (uint32_t)(K - k0), (uint32_t)(N - bn),
                         (uint32_t)K, BK, BN);
    }
    __builtin_amdgcn_s_wait_tensorcnt(0);
    __syncthreads();

    bf16x16 afrag[2], bfrag[4];
#pragma unroll
    for (int mt = 0; mt < 2; ++mt) {
      const int row = wm + mt * 16 + lm;
      V16 u;
      u.h[0] = *(const bf16x8*)&As[row][lh * 8];        // K = lh*8 + 0..7
      u.h[1] = *(const bf16x8*)&As[row][16 + lh * 8];   // K = 16+lh*8 + 0..7
      afrag[mt] = u.v;
    }
#pragma unroll
    for (int nt = 0; nt < 4; ++nt) {
      const int col = wn + nt * 16 + lm;
      V16 u;
      u.h[0] = *(const bf16x8*)&Bs[col][lh * 8];        // K = lh*8 + 0..7
      u.h[1] = *(const bf16x8*)&Bs[col][16 + lh * 8];   // K = 16+lh*8 + 0..7
      bfrag[nt] = u.v;
    }
#pragma unroll
    for (int mt = 0; mt < 2; ++mt)
#pragma unroll
      for (int nt = 0; nt < 4; ++nt)
        acc[mt][nt] = wmma_bf16(afrag[mt], bfrag[nt], acc[mt][nt]);
    __syncthreads();
  }

#pragma unroll
  for (int mt = 0; mt < 2; ++mt) {
#pragma unroll
    for (int nt = 0; nt < 4; ++nt) {
      const int col = bn + wn + nt * 16 + lm;
      const float bv = bias ? bias[col] : 0.f;
#pragma unroll
      for (int i = 0; i < 8; ++i) {
        const int row = bm + wm + mt * 16 + lh * 8 + i;
        C[(size_t)row * N + col] = acc[mt][nt][i] + bv;
      }
    }
  }
}

// ---------------------------------------------------------------- flash attention
// Q,K,V: bf16 [T, H*HD] (head h at columns h*128..). O: f32 [T, H*HD].
// Block = (query block of 128) x (one head); 8 waves, 16 query rows each.
__global__ __launch_bounds__(256) void flash_attn_kernel(
    const bf16* __restrict__ Q, const bf16* __restrict__ Kx,
    const bf16* __restrict__ Vx, float* __restrict__ O)
{
  const int h   = blockIdx.y;
  const int qb0 = blockIdx.x * 128;
  const int tid = threadIdx.x, lane = tid & 31, wid = tid >> 5;
  const int lh  = lane >> 4, lm = lane & 15;
  const int q0  = qb0 + wid * 16;
  const int ldq = D_MODEL;

  __shared__ bf16 Ks[32][HEAD_DIM + 8];   // [key][hd]
  __shared__ bf16 Vs[HEAD_DIM][32 + 8];   // transposed: [hd][key]
  __shared__ bf16 Ps[8][16][32 + 8];      // per-wave P scratch (C->A relayout)

  // Q fragments (16 rows x K=128) held in registers for the whole kernel.
  bf16x16 qf[4];
  {
    const bf16* qrow = Q + (size_t)(q0 + lm) * ldq + h * HEAD_DIM;
#pragma unroll
    for (int kk = 0; kk < 4; ++kk) {
      V16 u;
      u.h[0] = *(const bf16x8*)(qrow + kk * 32 + lh * 8);
      u.h[1] = *(const bf16x8*)(qrow + kk * 32 + 16 + lh * 8);
      qf[kk] = u.v;
    }
  }

  float m_i[8], l_i[8];
  f32x8 o_acc[8] = {};
#pragma unroll
  for (int i = 0; i < 8; ++i) { m_i[i] = -1e30f; l_i[i] = 0.f; }

  const int skey = tid >> 3;         // 0..31
  const int scg  = (tid & 7) * 16;   // 0..112
  const uint32_t lds_k = (uint32_t)(uintptr_t)&Ks[skey][scg];

  for (int kc = 0; kc < qb0 + 128; kc += 32) {
    // ---- K chunk: async global->LDS copy (no VGPR round trip)
    {
      const bf16* ksrc = Kx + (size_t)(kc + skey) * ldq + h * HEAD_DIM + scg;
      asm volatile("global_load_async_to_lds_b128 %0, %1, off"
                   :: "v"(lds_k), "v"((uint64_t)(uintptr_t)ksrc) : "memory");
      asm volatile("global_load_async_to_lds_b128 %0, %1, off"
                   :: "v"(lds_k + 16u), "v"((uint64_t)(uintptr_t)(ksrc + 8))
                   : "memory");
    }
    // ---- V chunk: load + transposed store into LDS
    {
      const bf16* vsrc = Vx + (size_t)(kc + skey) * ldq + h * HEAD_DIM + scg;
      bf16x8 lo = *(const bf16x8*)vsrc, hi = *(const bf16x8*)(vsrc + 8);
#pragma unroll
      for (int e = 0; e < 8; ++e) {
        Vs[scg + e][skey]     = lo[e];
        Vs[scg + 8 + e][skey] = hi[e];
      }
      // prefetch next chunk (global_prefetch_b8)
      __builtin_prefetch(vsrc + 32 * ldq, 0, 0);
      __builtin_prefetch(Kx + (size_t)(kc + 32 + skey) * ldq + h * HEAD_DIM + scg, 0, 0);
    }
    asm volatile("s_wait_asynccnt 0x0" ::: "memory");
    __syncthreads();

    if (kc <= q0 + 15) {   // uniform per wave: chunk intersects causal range
      // ---- S = Q @ K^T (two 16x16 tiles, K accumulated over HD=128)
      f32x8 s0 = {}, s1 = {};
#pragma unroll
      for (int kk = 0; kk < 4; ++kk) {
        V16 u0, u1;
        u0.h[0] = *(const bf16x8*)&Ks[lm][kk * 32 + lh * 8];
        u0.h[1] = *(const bf16x8*)&Ks[lm][kk * 32 + 16 + lh * 8];
        u1.h[0] = *(const bf16x8*)&Ks[16 + lm][kk * 32 + lh * 8];
        u1.h[1] = *(const bf16x8*)&Ks[16 + lm][kk * 32 + 16 + lh * 8];
        s0 = wmma_bf16(qf[kk], u0.v, s0);
        s1 = wmma_bf16(qf[kk], u1.v, s1);
      }

      // ---- online softmax (rows distributed: lane half lh, VGPR i -> row)
      const int col0 = kc + lm, col1 = kc + 16 + lm;
      float p0[8], p1[8];
#pragma unroll
      for (int i = 0; i < 8; ++i) {
        const int row = q0 + lh * 8 + i;
        float v0 = (col0 <= row) ? s0[i] * SM_SCALE : -1e30f;
        float v1 = (col1 <= row) ? s1[i] * SM_SCALE : -1e30f;
        float mx = fmaxf(v0, v1);
#pragma unroll
        for (int off = 8; off > 0; off >>= 1)
          mx = fmaxf(mx, __shfl_xor(mx, off, 32));
        const float nm = fmaxf(m_i[i], mx);
        const float e0 = __expf(v0 - nm);
        const float e1 = __expf(v1 - nm);
        float sm = e0 + e1;
#pragma unroll
        for (int off = 8; off > 0; off >>= 1)
          sm += __shfl_xor(sm, off, 32);
        const float alpha = __expf(m_i[i] - nm);
        m_i[i] = nm;
        l_i[i] = l_i[i] * alpha + sm;
        p0[i] = e0; p1[i] = e1;
#pragma unroll
        for (int nt = 0; nt < 8; ++nt) o_acc[nt][i] *= alpha;
      }

      // ---- relayout P (C-tile layout) into A-fragment layout via wave-local LDS
#pragma unroll
      for (int i = 0; i < 8; ++i) {
        Ps[wid][lh * 8 + i][lm]      = (bf16)p0[i];
        Ps[wid][lh * 8 + i][16 + lm] = (bf16)p1[i];
      }
      V16 pu;
      pu.h[0] = *(const bf16x8*)&Ps[wid][lm][lh * 8];
      pu.h[1] = *(const bf16x8*)&Ps[wid][lm][16 + lh * 8];
      const bf16x16 pf = pu.v;

      // ---- O += P @ V   (B-frags read from transposed Vs: contiguous)
#pragma unroll
      for (int nt = 0; nt < 8; ++nt) {
        V16 vu;
        vu.h[0] = *(const bf16x8*)&Vs[nt * 16 + lm][lh * 8];
        vu.h[1] = *(const bf16x8*)&Vs[nt * 16 + lm][16 + lh * 8];
        o_acc[nt] = wmma_bf16(pf, vu.v, o_acc[nt]);
      }
    }
    __syncthreads();
  }

#pragma unroll
  for (int i = 0; i < 8; ++i) {
    const float inv = 1.f / l_i[i];
    const size_t row = q0 + lh * 8 + i;
#pragma unroll
    for (int nt = 0; nt < 8; ++nt)
      O[row * ldq + h * HEAD_DIM + nt * 16 + lm] = o_acc[nt][i] * inv;
  }
}

// ---------------------------------------------------------------- elementwise
__global__ void cvt_f32_bf16(const float* __restrict__ in, bf16* __restrict__ out, int n) {
  int i = blockIdx.x * blockDim.x + threadIdx.x;
  if (i < n) out[i] = (bf16)in[i];
}

__global__ void cvt_strided_f32_bf16(const float* __restrict__ in, bf16* __restrict__ out,
                                     int rows, int cols, int ld_in) {
  int i = blockIdx.x * blockDim.x + threadIdx.x;
  if (i < rows * cols) {
    int r = i / cols, c = i - r * cols;
    out[i] = (bf16)in[(size_t)r * ld_in + c];
  }
}

// xt = kv_latent * softplus(t_scale) + t_shift, written as bf16
__global__ void splice_xt_kernel(const float* __restrict__ kvlat,
                                 const float* __restrict__ ts,
                                 const float* __restrict__ tsh,
                                 bf16* __restrict__ out, int n) {
  int i = blockIdx.x * blockDim.x + threadIdx.x;
  if (i < n) {
    int j = i % D_LAT;
    float s = ts[j];
    float sp = (s > 20.f) ? s : log1pf(__expf(s));
    out[i] = (bf16)(kvlat[i] * sp + tsh[j]);
  }
}

__global__ __launch_bounds__(256) void layernorm256_kernel(
    const float* __restrict__ in, const float* __restrict__ g,
    const float* __restrict__ b, bf16* __restrict__ out) {
  __shared__ float red[256];
  const int row = blockIdx.x, c = threadIdx.x;
  const float v = in[(size_t)row * D_CMP + c];
  red[c] = v; __syncthreads();
  for (int s = 128; s > 0; s >>= 1) { if (c < s) red[c] += red[c + s]; __syncthreads(); }
  const float mu = red[0] * (1.f / D_CMP);
  __syncthreads();
  const float d = v - mu;
  red[c] = d * d; __syncthreads();
  for (int s = 128; s > 0; s >>= 1) { if (c < s) red[c] += red[c + s]; __syncthreads(); }
  const float var = red[0] * (1.f / D_CMP);
  out[(size_t)row * D_CMP + c] = (bf16)(d * rsqrtf(var + 1e-5f) * g[c] + b[c]);
}

// RoPE on per-head 128-dim slices, f32 in (leading dim ld_in) -> bf16 out [T, 2048]
__global__ void rope_to_bf16_kernel(const float* __restrict__ in,
                                    bf16* __restrict__ out, int ld_in) {
  int i = blockIdx.x * blockDim.x + threadIdx.x;
  if (i >= T_TOK * N_HEADS * 64) return;
  const int d  = i % 64;
  const int hh = (i / 64) % N_HEADS;
  const int t  = i / (64 * N_HEADS);
  const float inv_freq = __expf(-(float)(2 * d) * (1.f / HEAD_DIM) * 9.210340371976184f);
  const float ang = (float)t * inv_freq;
  const float cv = __cosf(ang), sv = __sinf(ang);
  const size_t ib = (size_t)t * ld_in + hh * HEAD_DIM;
  const size_t ob = (size_t)t * D_MODEL + hh * HEAD_DIM;
  const float x1 = in[ib + d], x2 = in[ib + d + 64];
  out[ob + d]      = (bf16)(x1 * cv - x2 * sv);
  out[ob + d + 64] = (bf16)(x2 * cv + x1 * sv);
}

// ---------------------------------------------------------------- launch
extern "C" void kernel_launch(void* const* d_in, const int* in_sizes, int n_in,
                              void* d_out, int out_size, void* d_ws, size_t ws_size,
                              hipStream_t stream) {
  (void)in_sizes; (void)n_in; (void)out_size; (void)ws_size;
  const float* x     = (const float*)d_in[0];
  const float* Wq    = (const float*)d_in[1];
  const float* bq    = (const float*)d_in[2];
  const float* Wkl   = (const float*)d_in[3];
  const float* bkl   = (const float*)d_in[4];
  const float* tsc   = (const float*)d_in[5];
  const float* tsh   = (const float*)d_in[6];
  const float* Wc    = (const float*)d_in[7];
  const float* We    = (const float*)d_in[8];
  const float* lng   = (const float*)d_in[9];
  const float* lnb   = (const float*)d_in[10];
  const float* Wfrom = (const float*)d_in[11];
  const float* bfrom = (const float*)d_in[12];
  const float* Wout  = (const float*)d_in[13];
  const float* bout  = (const float*)d_in[14];

  char* base = (char*)d_ws;
  size_t off = 0;
  auto alloc = [&](size_t bytes) -> void* {
    void* r = base + off;
    off = (off + bytes + 255) & ~(size_t)255;
    return r;
  };

  // bf16 buffers
  bf16* xb      = (bf16*)alloc((size_t)T_TOK * D_MODEL * 2);
  bf16* wqb     = (bf16*)alloc((size_t)D_MODEL * D_MODEL * 2);
  bf16* wklb    = (bf16*)alloc((size_t)D_LAT * D_MODEL * 2);
  bf16* wcb     = (bf16*)alloc((size_t)D_CMP * D_LAT * 2);
  bf16* web     = (bf16*)alloc((size_t)D_LAT * D_CMP * 2);
  bf16* wfromb  = (bf16*)alloc((size_t)N_KV * D_LAT * 2);
  bf16* woutb   = (bf16*)alloc((size_t)D_MODEL * D_MODEL * 2);
  bf16* xtb     = (bf16*)alloc((size_t)T_TOK * D_LAT * 2);
  bf16* compb   = (bf16*)alloc((size_t)T_TOK * D_CMP * 2);
  bf16* kvlatpb = (bf16*)alloc((size_t)T_TOK * D_LAT * 2);
  bf16* qb      = (bf16*)alloc((size_t)T_TOK * D_MODEL * 2);
  bf16* kb      = (bf16*)alloc((size_t)T_TOK * D_MODEL * 2);
  bf16* vb      = (bf16*)alloc((size_t)T_TOK * D_MODEL * 2);
  bf16* attnb   = (bf16*)alloc((size_t)T_TOK * D_MODEL * 2);
  // f32 buffers
  float* qf     = (float*)alloc((size_t)T_TOK * D_MODEL * 4);
  float* kvlat  = (float*)alloc((size_t)T_TOK * D_LAT * 4);
  float* compf  = (float*)alloc((size_t)T_TOK * D_CMP * 4);
  float* kvlatp = (float*)alloc((size_t)T_TOK * D_LAT * 4);
  float* kvf    = (float*)alloc((size_t)T_TOK * N_KV * 4);
  float* attnf  = (float*)alloc((size_t)T_TOK * D_MODEL * 4);

  auto cvt = [&](const float* in, bf16* out, int n) {
    cvt_f32_bf16<<<(n + 255) / 256, 256, 0, stream>>>(in, out, n);
  };
  auto gemm = [&](const bf16* A, const bf16* B, const float* bias, float* C,
                  int M, int N, int K) {
    gemm_bf16_kernel<<<dim3(N / BN, M / BM), 256, 0, stream>>>(A, B, bias, C, M, N, K);
  };

  // ---- convert activations + weights to bf16
  cvt(x,     xb,     T_TOK * D_MODEL);
  cvt(Wq,    wqb,    D_MODEL * D_MODEL);
  cvt(Wkl,   wklb,   D_LAT * D_MODEL);
  cvt(Wc,    wcb,    D_CMP * D_LAT);
  cvt(We,    web,    D_LAT * D_CMP);
  cvt(Wfrom, wfromb, N_KV * D_LAT);
  cvt(Wout,  woutb,  D_MODEL * D_MODEL);

  // ---- Q projection: [T,2048] = x @ Wq^T + bq
  gemm(xb, wqb, bq, qf, T_TOK, D_MODEL, D_MODEL);

  // ---- KV latent + KVSplice bottleneck
  gemm(xb, wklb, bkl, kvlat, T_TOK, D_LAT, D_MODEL);
  splice_xt_kernel<<<(T_TOK * D_LAT + 255) / 256, 256, 0, stream>>>(
      kvlat, tsc, tsh, xtb, T_TOK * D_LAT);
  gemm(xtb, wcb, nullptr, compf, T_TOK, D_CMP, D_LAT);
  layernorm256_kernel<<<T_TOK, 256, 0, stream>>>(compf, lng, lnb, compb);
  gemm(compb, web, nullptr, kvlatp, T_TOK, D_LAT, D_CMP);
  cvt(kvlatp, kvlatpb, T_TOK * D_LAT);

  // ---- decompress latent to K,V: [T,4096] = latent' @ Wfrom^T + bfrom
  gemm(kvlatpb, wfromb, bfrom, kvf, T_TOK, N_KV, D_LAT);

  // ---- RoPE(Q), RoPE(K) -> bf16; V -> bf16
  {
    const int n = T_TOK * N_HEADS * 64;
    rope_to_bf16_kernel<<<(n + 255) / 256, 256, 0, stream>>>(qf, qb, D_MODEL);
    rope_to_bf16_kernel<<<(n + 255) / 256, 256, 0, stream>>>(kvf, kb, N_KV);
    cvt_strided_f32_bf16<<<(T_TOK * D_MODEL + 255) / 256, 256, 0, stream>>>(
        kvf + D_MODEL, vb, T_TOK, D_MODEL, N_KV);
  }

  // ---- causal flash attention
  flash_attn_kernel<<<dim3(T_TOK / 128, N_HEADS), 256, 0, stream>>>(qb, kb, vb, attnf);

  // ---- output projection: d_out = attn @ Wout^T + bout
  cvt(attnf, attnb, T_TOK * D_MODEL);
  gemm(attnb, woutb, bout, (float*)d_out, T_TOK, D_MODEL, D_MODEL);
}
